// Model_2680059592777
// MI455X (gfx1250) — matrix operand order repaired and checked by reading, not compiled, not run on hardware
//
#include <hip/hip_runtime.h>
#include <math.h>

typedef __attribute__((ext_vector_type(16))) __bf16 v16bf;
typedef __attribute__((ext_vector_type(8)))  float  v8f;

static constexpr int NC = 5000, EC = 40000, NT = 32, ET = 256;
static constexpr int H = 256, RH = 128, G3 = 384; // 3*RH

__device__ __forceinline__ unsigned short f2bf(float x) {
  unsigned u = __builtin_bit_cast(unsigned, x);
  return (unsigned short)((u + 0x7FFFu + ((u >> 16) & 1u)) >> 16);
}
__device__ __forceinline__ float sigmoidf(float x) { return 1.f / (1.f + expf(-x)); }

// ---------------------------------------------------------------------------
// Generic GEMM: C[M,N] = A[M,K] @ B[K,N]   (fp32 in/out, bf16 WMMA compute)
// Block tile 128x128, K-step 32, 256 threads = 8 waves (4x2 wave grid).
// LDS staging is pre-swizzled into WMMA fragment layouts:
//   A 16x32 tile:  lane = m + 16*((k%16)/8), half = (k/16)*8 + (k%8)
//   B 32x16 tile:  lane = n + 16*((k%16)/8), half = (k/16)*8 + (k%8)
// so each fragment load is a contiguous 32B v16bf read.
// ---------------------------------------------------------------------------
__global__ __launch_bounds__(256) void k_gemm(
    const float* __restrict__ A, const float* __restrict__ B,
    float* __restrict__ C, int M, int N, int K,
    int lda, int ldb, int ldc, int accumulate) {
  __shared__ alignas(32) unsigned short aS[8 * 32 * 16];
  __shared__ alignas(32) unsigned short bS[8 * 32 * 16];
  const int tid = threadIdx.x;
  const int w = tid >> 5, lane = tid & 31;
  const int mw = w >> 1, nw = w & 1;
  const int m0 = blockIdx.x * 128, n0 = blockIdx.y * 128;

  v8f acc[2][4];
  for (int i = 0; i < 2; ++i)
    for (int j = 0; j < 4; ++j)
      for (int r = 0; r < 8; ++r) acc[i][j][r] = 0.f;

  for (int k0 = 0; k0 < K; k0 += 32) {
    for (int i = tid; i < 4096; i += 256) {
      int mm = i >> 5, kk = i & 31;
      int row = m0 + mm, kg = k0 + kk;
      float v = (row < M && kg < K) ? A[(size_t)row * lda + kg] : 0.f;
      int mt = mm >> 4, ml = mm & 15;
      int klo = kk & 15, hi = kk >> 4;
      int al = ml + 16 * ((klo >> 3) & 1);
      int hf = hi * 8 + (klo & 7);
      aS[mt * 512 + al * 16 + hf] = f2bf(v);
    }
    for (int i = tid; i < 4096; i += 256) {
      int kk = i >> 7, nn = i & 127;
      int col = n0 + nn, kg = k0 + kk;
      float v = (col < N && kg < K) ? B[(size_t)kg * ldb + col] : 0.f;
      int nt = nn >> 4, nl = nn & 15;
      int bl = nl + 16 * ((kk >> 3) & 1);
      int hf = (kk >> 4) * 8 + (kk & 7);
      bS[nt * 512 + bl * 16 + hf] = f2bf(v);
    }
    __syncthreads();
    v16bf af0 = *(const v16bf*)(aS + (mw * 2 + 0) * 512 + lane * 16);
    v16bf af1 = *(const v16bf*)(aS + (mw * 2 + 1) * 512 + lane * 16);
    for (int j = 0; j < 4; ++j) {
      v16bf bfj = *(const v16bf*)(bS + (nw * 4 + j) * 512 + lane * 16);
      acc[0][j] = __builtin_amdgcn_wmma_f32_16x16x32_bf16(
          false, af0, false, bfj, (short)0, acc[0][j], false, false);
      acc[1][j] = __builtin_amdgcn_wmma_f32_16x16x32_bf16(
          false, af1, false, bfj, (short)0, acc[1][j], false, false);
    }
    __syncthreads();
  }
  const int mofs = (lane >> 4) * 8, nofs = lane & 15;
  for (int i = 0; i < 2; ++i)
    for (int j = 0; j < 4; ++j)
      for (int r = 0; r < 8; ++r) {
        int row = m0 + (mw * 2 + i) * 16 + r + mofs;
        int col = n0 + (nw * 4 + j) * 16 + nofs;
        if (row < M && col < N) {
          size_t idx = (size_t)row * ldc + col;
          float v = acc[i][j][r];
          C[idx] = accumulate ? (C[idx] + v) : v;
        }
      }
}

// ---------------------------------------------------------------------------
// GRU scan (Keras reset_after). Block owns 64 batch rows; rk resident in LDS
// (bf16, fragment-swizzled, 96KB); h state kept in registers (D-layout) and
// mirrored to LDS in A-fragment layout for the next step's h@rk WMMAs.
// Wave w: m-tile mt=w/2 (16 rows), column half ch=w&1 (64 of 128 hidden cols),
// owning the full z/r/hh triple (12 accumulator tiles) for those columns.
// mode==1: xz = xc[b] + xt[t] (layer-1 broadcast decomposition, xt holds b0)
// mode==0: xz = precomputed [B*T, 384] (b0 already added)
// ---------------------------------------------------------------------------
__global__ __launch_bounds__(256) void k_gru_scan(
    const float* __restrict__ xz, const float* __restrict__ xc,
    const float* __restrict__ xt, const float* __restrict__ rk,
    const float* __restrict__ rb, float* __restrict__ ys,
    int B, int T, int reverse, int dirofs, int mode) {
  __shared__ alignas(32) unsigned short rkS[4 * 24 * 32 * 16]; // 96 KB
  __shared__ alignas(32) unsigned short hS[4 * 4 * 32 * 16];   // 16 KB
  const int tid = threadIdx.x;
  const int w = tid >> 5, lane = tid & 31;
  const int mt = w >> 1, ch = w & 1;
  const int row0 = blockIdx.x * 64;

  for (int i = tid; i < 128 * 384; i += 256) {
    int k = i / 384, n = i % 384;
    float v = rk[i];
    int kc = k >> 5, kl = k & 31;
    int nt = n >> 4, nl = n & 15;
    int bl = nl + 16 * ((kl >> 3) & 1);
    int hf = (kl >> 4) * 8 + (kl & 7);
    rkS[(kc * 24 + nt) * 512 + bl * 16 + hf] = f2bf(v);
  }
  for (int i = tid; i < 4 * 4 * 32 * 16; i += 256) hS[i] = 0;

  v8f hprev[4];
  for (int j = 0; j < 4; ++j)
    for (int r = 0; r < 8; ++r) hprev[j][r] = 0.f;

  float rbz[4], rbr[4], rbh[4];
  for (int j = 0; j < 4; ++j) {
    int c = (lane & 15) + 16 * (ch * 4 + j);
    rbz[j] = rb[c]; rbr[j] = rb[128 + c]; rbh[j] = rb[256 + c];
  }
  __syncthreads();

  const int mofs = (lane >> 4) * 8;
  for (int s = 0; s < T; ++s) {
    int t = reverse ? (T - 1 - s) : s;
    v16bf af[4];
    for (int kc = 0; kc < 4; ++kc)
      af[kc] = *(const v16bf*)(hS + (mt * 4 + kc) * 512 + lane * 16);
    v8f az[4], ar[4], ah[4];
    for (int j = 0; j < 4; ++j)
      for (int r = 0; r < 8; ++r) { az[j][r] = 0.f; ar[j][r] = 0.f; ah[j][r] = 0.f; }
    for (int j = 0; j < 4; ++j) {
      int ntz = ch * 4 + j, ntr = 8 + ch * 4 + j, nth = 16 + ch * 4 + j;
      for (int kc = 0; kc < 4; ++kc) {
        v16bf bz = *(const v16bf*)(rkS + (kc * 24 + ntz) * 512 + lane * 16);
        az[j] = __builtin_amdgcn_wmma_f32_16x16x32_bf16(
            false, af[kc], false, bz, (short)0, az[j], false, false);
        v16bf br = *(const v16bf*)(rkS + (kc * 24 + ntr) * 512 + lane * 16);
        ar[j] = __builtin_amdgcn_wmma_f32_16x16x32_bf16(
            false, af[kc], false, br, (short)0, ar[j], false, false);
        v16bf bh = *(const v16bf*)(rkS + (kc * 24 + nth) * 512 + lane * 16);
        ah[j] = __builtin_amdgcn_wmma_f32_16x16x32_bf16(
            false, af[kc], false, bh, (short)0, ah[j], false, false);
      }
    }
    __syncthreads(); // hS fully consumed by all waves before rewrite
    for (int j = 0; j < 4; ++j) {
      int c = (lane & 15) + 16 * (ch * 4 + j);
      for (int r = 0; r < 8; ++r) {
        int bRow = row0 + mt * 16 + r + mofs;
        bool valid = bRow < B;
        float xzz = 0.f, xzr = 0.f, xzh = 0.f;
        if (valid) {
          if (mode == 1) {
            const float* pc = xc + (size_t)bRow * 384;
            const float* pt = xt + (size_t)t * 384;
            xzz = pc[c] + pt[c];
            xzr = pc[128 + c] + pt[128 + c];
            xzh = pc[256 + c] + pt[256 + c];
          } else {
            const float* p = xz + ((size_t)bRow * T + t) * 384;
            xzz = p[c]; xzr = p[128 + c]; xzh = p[256 + c];
          }
        }
        float iz = az[j][r] + rbz[j];
        float ir = ar[j][r] + rbr[j];
        float ih = ah[j][r] + rbh[j];
        float zg = sigmoidf(xzz + iz);
        float rg = sigmoidf(xzr + ir);
        float hg = tanhf(xzh + rg * ih);
        float hn = zg * hprev[j][r] + (1.f - zg) * hg;
        hprev[j][r] = hn;
        if (valid) ys[((size_t)bRow * T + t) * 256 + dirofs + c] = hn;
        int mm = r + mofs;
        int kc2 = c >> 5, kl = c & 31;
        int klo = kl & 15, hi = kl >> 4;
        int al = mm + 16 * ((klo >> 3) & 1);
        int hf = hi * 8 + (klo & 7);
        hS[(mt * 4 + kc2) * 512 + al * 16 + hf] = f2bf(hn);
      }
    }
    __syncthreads();
  }
}

// ------------------------- small elementwise kernels -----------------------
__global__ void k_deg(const int* __restrict__ src, const int* __restrict__ dst,
                      float* dego, float* degi, int E) {
  int e = blockIdx.x * blockDim.x + threadIdx.x;
  if (e < E) { atomicAdd(dego + src[e], 1.f); atomicAdd(degi + dst[e], 1.f); }
}
__global__ void k_rsqrt_clip(float* d, int n) {
  int i = blockIdx.x * blockDim.x + threadIdx.x;
  if (i < n) { float v = d[i]; d[i] = rsqrtf(v < 1.f ? 1.f : v); }
}
__global__ void k_build_cx0(const float* __restrict__ cf, const float* __restrict__ emb,
                            const int* __restrict__ ty, float* out, int n) {
  int i = blockIdx.x * blockDim.x + threadIdx.x;
  if (i < n * 20) {
    int r = i / 20, c = i % 20;
    out[i] = (c < 16) ? cf[r * 16 + c] : emb[ty[r] * 4 + (c - 16)];
  }
}
__global__ void k_scale(const float* __restrict__ x, const float* __restrict__ dsc,
                        float* h, int rows, int cols) {
  size_t i = (size_t)blockIdx.x * blockDim.x + threadIdx.x;
  if (i < (size_t)rows * cols) h[i] = x[i] * dsc[i / cols];
}
__global__ void k_scatter(const float* __restrict__ h, const int* __restrict__ src,
                          const int* __restrict__ dst, float* agg, int E, int cols) {
  size_t i = (size_t)blockIdx.x * blockDim.x + threadIdx.x;
  if (i < (size_t)E * cols) {
    int e = (int)(i / cols), f = (int)(i % cols);
    atomicAdd(agg + (size_t)dst[e] * cols + f, h[(size_t)src[e] * cols + f]);
  }
}
__global__ void k_scatter_e(const float* __restrict__ ef, const int* __restrict__ dst,
                            float* agg, int E, int cols) {
  size_t i = (size_t)blockIdx.x * blockDim.x + threadIdx.x;
  if (i < (size_t)E * cols) {
    int e = (int)(i / cols), f = (int)(i % cols);
    atomicAdd(agg + (size_t)dst[e] * cols + f, ef[i]);
  }
}
__global__ void k_gconv_post(const float* __restrict__ rst, const float* __restrict__ agge,
                             const float* __restrict__ W2, const float* __restrict__ bias,
                             const float* __restrict__ dsci, const float* __restrict__ h,
                             float* out, int rows, int cols, int act, int residual) {
  size_t i = (size_t)blockIdx.x * blockDim.x + threadIdx.x;
  if (i < (size_t)rows * cols) {
    int r = (int)(i / cols), c = (int)(i % cols);
    float e0 = agge[r * 4 + 0] * W2[0 * cols + c] + agge[r * 4 + 1] * W2[1 * cols + c] +
               agge[r * 4 + 2] * W2[2 * cols + c] + agge[r * 4 + 3] * W2[3 * cols + c];
    float v = (rst[i] + e0) * dsci[r] + bias[c];
    if (act) v = sigmoidf(v);
    if (residual) v += h[i];
    out[i] = v;
  }
}
__global__ void k_add_bias(float* x, const float* __restrict__ b, long long rows, int cols) {
  size_t i = (size_t)blockIdx.x * blockDim.x + threadIdx.x;
  if (i < (size_t)rows * cols) x[i] += b[i % cols];
}
__global__ void k_head(const float* __restrict__ ys, const float* __restrict__ Wf,
                       const float* __restrict__ bf, float* out, int rows) {
  int r = blockIdx.x * blockDim.x + threadIdx.x;
  if (r < rows) {
    const float* p = ys + (size_t)r * 256;
    float a0 = bf[0], a1 = bf[1];
    for (int k = 0; k < 256; ++k) { float v = p[k]; a0 += v * Wf[2 * k]; a1 += v * Wf[2 * k + 1]; }
    float m = fmaxf(a0, a1);
    float lse = m + logf(expf(a0 - m) + expf(a1 - m));
    out[2 * r] = a0 - lse;
    out[2 * r + 1] = a1 - lse;
  }
}

// ---------------------------------------------------------------------------
extern "C" void kernel_launch(void* const* d_in, const int* in_sizes, int n_in,
                              void* d_out, int out_size, void* d_ws, size_t ws_size,
                              hipStream_t stream) {
  (void)in_sizes; (void)n_in; (void)out_size; (void)ws_size;
  // setup_inputs() insertion order, params flattened recursively in insertion order
  const float* cfeats = (const float*)d_in[0];
  const float* cef    = (const float*)d_in[1];
  const int*   ctypes = (const int*)d_in[2];
  const float* tfeats = (const float*)d_in[3];
  const float* tef    = (const float*)d_in[4];
  const int*   csrc   = (const int*)d_in[6];
  const int*   cdst   = (const int*)d_in[7];
  const int*   tsrc   = (const int*)d_in[8];
  const int*   tdst   = (const int*)d_in[9];
  const float* op_emb = (const float*)d_in[10];
  const float *cW[6], *cb[6], *tW[4], *tb[4];
  for (int l = 0; l < 6; ++l) { cW[l] = (const float*)d_in[11 + l]; cb[l] = (const float*)d_in[17 + l]; }
  for (int l = 0; l < 4; ++l) { tW[l] = (const float*)d_in[23 + l]; tb[l] = (const float*)d_in[27 + l]; }
  // gru: [31..36] layer0 (k,rk,b)x2 dirs, [37..42] layer1
  const float* Wf = (const float*)d_in[43];
  const float* bfp = (const float*)d_in[44];

  char* wp = (char*)d_ws;
  auto alloc = [&](size_t nfloats) -> float* {
    float* p = (float*)wp;
    wp += ((nfloats * sizeof(float) + 255) / 256) * 256;
    return p;
  };
  float* cx0   = alloc((size_t)NC * 20);
  float* cdego = alloc(NC);
  float* cdegi = alloc(NC);
  float* tdego = alloc(NT);
  float* tdegi = alloc(NT);
  float* cagge = alloc((size_t)NC * 4);
  float* tagge = alloc((size_t)NT * 4);
  float* chb   = alloc((size_t)NC * H);
  float* cagg  = alloc((size_t)NC * H);
  float* crst  = alloc((size_t)NC * H);
  float* cxA   = alloc((size_t)NC * H);
  float* cxB   = alloc((size_t)NC * H);
  float* thb   = alloc((size_t)NT * H);
  float* tagg  = alloc((size_t)NT * H);
  float* trst  = alloc((size_t)NT * H);
  float* txA   = alloc((size_t)NT * H);
  float* txB   = alloc((size_t)NT * H);
  float* xc    = alloc((size_t)NC * G3);
  float* xtb   = alloc((size_t)NT * G3);
  float* ys1   = alloc((size_t)NC * NT * H);
  float* xz2   = alloc((size_t)NC * NT * G3);
  float* ys2   = alloc((size_t)NC * NT * H);

  // ---- degrees + edge aggregates (layer-invariant) ----
  hipMemsetAsync(cdego, 0, NC * sizeof(float), stream);
  hipMemsetAsync(cdegi, 0, NC * sizeof(float), stream);
  hipMemsetAsync(tdego, 0, NT * sizeof(float), stream);
  hipMemsetAsync(tdegi, 0, NT * sizeof(float), stream);
  k_deg<<<(EC + 255) / 256, 256, 0, stream>>>(csrc, cdst, cdego, cdegi, EC);
  k_deg<<<(ET + 255) / 256, 256, 0, stream>>>(tsrc, tdst, tdego, tdegi, ET);
  k_rsqrt_clip<<<(NC + 255) / 256, 256, 0, stream>>>(cdego, NC);
  k_rsqrt_clip<<<(NC + 255) / 256, 256, 0, stream>>>(cdegi, NC);
  k_rsqrt_clip<<<1, 256, 0, stream>>>(tdego, NT);
  k_rsqrt_clip<<<1, 256, 0, stream>>>(tdegi, NT);
  hipMemsetAsync(cagge, 0, (size_t)NC * 4 * sizeof(float), stream);
  hipMemsetAsync(tagge, 0, (size_t)NT * 4 * sizeof(float), stream);
  k_scatter_e<<<((size_t)EC * 4 + 255) / 256, 256, 0, stream>>>(cef, cdst, cagge, EC, 4);
  k_scatter_e<<<((size_t)ET * 4 + 255) / 256, 256, 0, stream>>>(tef, tdst, tagge, ET, 4);
  k_build_cx0<<<((size_t)NC * 20 + 255) / 256, 256, 0, stream>>>(cfeats, op_emb, ctypes, cx0, NC);

  // ---- circuit graph convs ----
  {
    const int Fl[6]  = {20, 256, 256, 256, 256, 256};
    const int res[6] = {0, 1, 1, 1, 1, 0};
    const int act[6] = {1, 1, 1, 1, 1, 0};
    const float* cur = cx0;
    float* nxt = cxA; float* other = cxB;
    for (int l = 0; l < 6; ++l) {
      int F = Fl[l];
      k_scale<<<((size_t)NC * F + 255) / 256, 256, 0, stream>>>(cur, cdego, chb, NC, F);
      hipMemsetAsync(cagg, 0, (size_t)NC * F * sizeof(float), stream);
      k_scatter<<<((size_t)EC * F + 255) / 256, 256, 0, stream>>>(chb, csrc, cdst, cagg, EC, F);
      dim3 gg((NC + 127) / 128, (H + 127) / 128);
      k_gemm<<<gg, 256, 0, stream>>>(cagg, cW[l], crst, NC, H, F, F, H, H, 0);
      k_gconv_post<<<((size_t)NC * H + 255) / 256, 256, 0, stream>>>(
          crst, cagge, cW[l] + (size_t)F * H, cb[l], cdegi, chb, nxt, NC, H, act[l], res[l]);
      cur = nxt; float* tp = nxt; nxt = other; other = tp;
    }
    // c_emb == cur
    const float* c_emb = cur;

    // ---- template graph convs ----
    const int tFl[4]  = {16, 256, 256, 256};
    const int tres[4] = {0, 1, 1, 0};
    const int tact[4] = {1, 1, 1, 0};
    const float* tcur = tfeats;
    float* tnxt = txA; float* tother = txB;
    for (int l = 0; l < 4; ++l) {
      int F = tFl[l];
      k_scale<<<((size_t)NT * F + 255) / 256, 256, 0, stream>>>(tcur, tdego, thb, NT, F);
      hipMemsetAsync(tagg, 0, (size_t)NT * F * sizeof(float), stream);
      k_scatter<<<((size_t)ET * F + 255) / 256, 256, 0, stream>>>(thb, tsrc, tdst, tagg, ET, F);
      dim3 gg((NT + 127) / 128, (H + 127) / 128);
      k_gemm<<<gg, 256, 0, stream>>>(tagg, tW[l], trst, NT, H, F, F, H, H, 0);
      k_gconv_post<<<((size_t)NT * H + 255) / 256, 256, 0, stream>>>(
          trst, tagge, tW[l] + (size_t)F * H, tb[l], tdegi, thb, tnxt, NT, H, tact[l], tres[l]);
      tcur = tnxt; float* tp = tnxt; tnxt = tother; tother = tp;
    }
    const float* t_emb = tcur;

    // ---- biGRU layer 1 (broadcast-decomposed input projection) ----
    for (int d = 0; d < 2; ++d) {
      const float* gk  = (const float*)d_in[31 + 3 * d];
      const float* grk = (const float*)d_in[31 + 3 * d + 1];
      const float* gb  = (const float*)d_in[31 + 3 * d + 2]; // [2,384]
      dim3 g1((NC + 127) / 128, (G3 + 127) / 128);
      k_gemm<<<g1, 256, 0, stream>>>(c_emb, gk, xc, NC, G3, H, H, G3, G3, 0);
      dim3 g2((NT + 127) / 128, (G3 + 127) / 128);
      k_gemm<<<g2, 256, 0, stream>>>(t_emb, gk + (size_t)H * G3, xtb, NT, G3, H, H, G3, G3, 0);
      k_add_bias<<<((size_t)NT * G3 + 255) / 256, 256, 0, stream>>>(xtb, gb, NT, G3);
      k_gru_scan<<<(NC + 63) / 64, 256, 0, stream>>>(
          nullptr, xc, xtb, grk, gb + G3, ys1, NC, NT, d, d * RH, 1);
    }

    // ---- biGRU layer 2 ----
    for (int d = 0; d < 2; ++d) {
      const float* gk  = (const float*)d_in[37 + 3 * d];
      const float* grk = (const float*)d_in[37 + 3 * d + 1];
      const float* gb  = (const float*)d_in[37 + 3 * d + 2];
      long long Mrows = (long long)NC * NT;
      dim3 g1((unsigned)((Mrows + 127) / 128), (G3 + 127) / 128);
      k_gemm<<<g1, 256, 0, stream>>>(ys1, gk, xz2, (int)Mrows, G3, H, H, G3, G3, 0);
      k_add_bias<<<(unsigned)(((size_t)Mrows * G3 + 255) / 256), 256, 0, stream>>>(xz2, gb, Mrows, G3);
      k_gru_scan<<<(NC + 63) / 64, 256, 0, stream>>>(
          xz2, nullptr, nullptr, grk, gb + G3, ys2, NC, NT, d, d * RH, 0);
    }

    // ---- head: [256]@[256,2] + log_softmax ----
    k_head<<<((NC * NT) + 255) / 256, 256, 0, stream>>>(ys2, Wf, bfp, (float*)d_out, NC * NT);
  }
}
